// EfficienctMultiHeadAttention_43782896616049
// MI455X (gfx1250) — matrix operand order repaired and checked by reading, not compiled, not run on hardware
//
#include <hip/hip_runtime.h>
#include <math.h>

typedef __attribute__((ext_vector_type(16))) _Float16 v16h;
typedef __attribute__((ext_vector_type(8)))  _Float16 h8;
typedef __attribute__((ext_vector_type(8)))  float    v8f;
typedef __attribute__((ext_vector_type(4)))  unsigned int u32x4;
typedef __attribute__((ext_vector_type(8)))  int      i32x8;
typedef __attribute__((ext_vector_type(4)))  int      i32x4;

constexpr int EMBED    = 1024;
constexpr int HEADS    = 16;
constexpr int HEAD_DIM = 64;
constexpr int BATCH    = 2;
constexpr int SEQ      = 2048;
constexpr int ROWS     = BATCH * SEQ;   // 4096
constexpr int E3       = 3 * EMBED;

// LDS layout for attention kernel (dynamic shared => descriptor offsets from 0)
constexpr unsigned LDS_K_OFF = 0;          // 2 buffers * 32*64 f16 = 8192 B
constexpr unsigned LDS_V_OFF = 8192;       // 2 buffers * 64*32 f16 = 8192 B
constexpr unsigned LDS_P_OFF = 16384;      // 16*34 f16 = 1088 B
constexpr unsigned LDS_TOTAL = 17472;

#if defined(__AMDGCN__) && __has_builtin(__builtin_amdgcn_tensor_load_to_lds) && \
    __has_builtin(__builtin_amdgcn_s_wait_tensorcnt)
#define USE_TDM 1
#else
#define USE_TDM 0
#endif

__device__ __forceinline__ v8f wmma_f16(const v16h& a, const v16h& b, const v8f& c) {
    return __builtin_amdgcn_wmma_f32_16x16x32_f16(
        false, a, false, b, (short)0, c, false, false);
}

#if USE_TDM
// Build a 2D TDM descriptor (data_size = 2 bytes) and issue TENSOR_LOAD_TO_LDS.
// dims/tiles in elements; stride0 in elements (48-bit field).
__device__ __forceinline__ void tdm_load_2d(const void* gptr, unsigned lds_off,
                                            unsigned dim0, unsigned dim1,
                                            unsigned t0, unsigned t1,
                                            unsigned long long stride0) {
    unsigned long long ga = (unsigned long long)(__UINTPTR_TYPE__)gptr;
    u32x4 g0;
    g0[0] = 1u;                                   // count=1, is_restore=0, gather=0
    g0[1] = lds_off;                              // lds_addr [63:32]
    g0[2] = (unsigned)(ga & 0xFFFFFFFFu);         // global_addr [95:64]
    g0[3] = (unsigned)((ga >> 32) & 0x1FFFFFFu)   // global_addr [120:96]
          | (2u << 30);                           // type=2 ("image") [127:126]
    i32x8 g1;
    g1[0] = (int)(1u << 16);                      // wg_mask=0, data_size=1 (2B)
    g1[1] = (int)((dim0 & 0xFFFFu) << 16);        // tensor_dim0[15:0] -> bits 63:48
    g1[2] = (int)((dim0 >> 16) | ((dim1 & 0xFFFFu) << 16)); // dim0[31:16], dim1[15:0]
    g1[3] = (int)((dim1 >> 16) | ((t0 & 0xFFFFu) << 16));   // dim1[31:16], tile_dim0
    g1[4] = (int)(t1 & 0xFFFFu);                  // tile_dim1 (tile_dim2 = 0)
    g1[5] = (int)(unsigned)(stride0 & 0xFFFFFFFFull);       // dim0_stride[31:0]
    g1[6] = (int)(unsigned)((stride0 >> 32) & 0xFFFFull);   // dim0_stride[47:32]
    g1[7] = 0;                                    // tensor_dim1_stride (unused, 2D)
    i32x4 z4 = {};
    i32x8 z8 = {};
    // 6-arg toolchain form: (g0, g1, g2, g3, <extra>, cpol)
    __builtin_amdgcn_tensor_load_to_lds(g0, g1, z4, z4, z8, 0);
}
#endif

// ---------------------------------------------------------------------------
// Staging passes
// ---------------------------------------------------------------------------
__global__ __launch_bounds__(256)
void cvt_f32_to_f16(const float* __restrict__ in, _Float16* __restrict__ out, int n) {
    int i = blockIdx.x * 256 + threadIdx.x;
    if (i < n) out[i] = (_Float16)in[i];
}

__global__ __launch_bounds__(256)
void transpose_cvt(const float* __restrict__ in, _Float16* __restrict__ out, int K, int N) {
    size_t i = (size_t)blockIdx.x * 256 + threadIdx.x;
    if (i < (size_t)K * N) {
        int k = (int)(i % K), n = (int)(i / K);
        out[(size_t)n * K + k] = (_Float16)in[(size_t)k * N + n];
    }
}

__global__ __launch_bounds__(256)
void build_vt(const _Float16* __restrict__ qkv_h, _Float16* __restrict__ vt) {
    size_t i = (size_t)blockIdx.x * 256 + threadIdx.x;
    if (i >= (size_t)BATCH * HEADS * HEAD_DIM * SEQ) return;
    int s = (int)(i % SEQ);
    int r = (int)(i / SEQ);
    int d = r % HEAD_DIM;  r /= HEAD_DIM;
    int h = r % HEADS;
    int b = r / HEADS;
    vt[i] = qkv_h[((size_t)b * SEQ + s) * E3 + 2 * EMBED + h * HEAD_DIM + d];
}

// ---------------------------------------------------------------------------
// C[M,N] = A[M,K] * Bt[N,K]^T + bias[N]   (f16 in, f16/f32 out)
// One wave computes a 32x64 tile; 8 WMMAs per 32-deep K-step.
// ---------------------------------------------------------------------------
template <typename OutT>
__global__ __launch_bounds__(32)
void gemm_bias_wmma(const _Float16* __restrict__ A, const _Float16* __restrict__ Bt,
                    const float* __restrict__ bias, OutT* __restrict__ C,
                    int M, int N, int K) {
    const int lane = threadIdx.x;
    const int row  = lane & 15;
    const int hi   = lane >> 4;
    const int m0   = blockIdx.y * 32;
    const int n0   = blockIdx.x * 64;

    v8f acc[2][4] = {};
    const _Float16* arow0 = A + (size_t)(m0 + row) * K;
    const _Float16* arow1 = A + (size_t)(m0 + 16 + row) * K;

    for (int k0 = 0; k0 < K; k0 += 32) {
        v16h a[2];
#pragma unroll
        for (int mt = 0; mt < 2; ++mt) {
            const _Float16* ar = (mt == 0 ? arow0 : arow1) + k0;
            h8 lo = *(const h8*)(ar + hi * 16);
            h8 hp = *(const h8*)(ar + hi * 16 + 8);
#pragma unroll
            for (int i = 0; i < 8; ++i) { a[mt][i] = lo[i]; a[mt][8 + i] = hp[i]; }
        }
        if (k0 + 32 < K) {
            __builtin_prefetch(arow0 + k0 + 32, 0, 1);
            __builtin_prefetch(arow1 + k0 + 32, 0, 1);
        }
#pragma unroll
        for (int g = 0; g < 4; ++g) {
            const int col = n0 + g * 16 + row;
            v16h bf = *(const v16h*)(Bt + (size_t)col * K + k0 + hi * 16);
            acc[0][g] = wmma_f16(a[0], bf, acc[0][g]);
            acc[1][g] = wmma_f16(a[1], bf, acc[1][g]);
        }
    }
#pragma unroll
    for (int mt = 0; mt < 2; ++mt)
#pragma unroll
        for (int g = 0; g < 4; ++g) {
            const int col = n0 + g * 16 + row;
            const float bv = bias[col];
#pragma unroll
            for (int r = 0; r < 8; ++r)
                C[(size_t)(m0 + mt * 16 + r + 8 * hi) * N + col] = (OutT)(acc[mt][g][r] + bv);
        }
}

// ---------------------------------------------------------------------------
// Flash attention with TDM double-buffered K/V tiles in LDS.
// grid = (S/16, HEADS, BATCH), block = 32, dynamic LDS = LDS_TOTAL bytes.
// ---------------------------------------------------------------------------
__global__ __launch_bounds__(32)
void flash_attn_wmma(const _Float16* __restrict__ qkv_h, const _Float16* __restrict__ vt,
                     _Float16* __restrict__ attn_h) {
    extern __shared__ char smem[];
    _Float16* ldsK  = (_Float16*)(smem + LDS_K_OFF);   // [buf][32 keys][64 d]
    _Float16* ldsV  = (_Float16*)(smem + LDS_V_OFF);   // [buf][64 d][32 keys]
    _Float16* lds_p = (_Float16*)(smem + LDS_P_OFF);   // [16][34]

    const int lane = threadIdx.x;
    const int row  = lane & 15;
    const int hi   = lane >> 4;
    const int qt = blockIdx.x, h = blockIdx.y, b = blockIdx.z;
    const size_t base = (size_t)b * SEQ;
    const float SM = 0.125f;   // 1/sqrt(64)

    const _Float16* kbase = qkv_h + base * E3 + EMBED + h * HEAD_DIM;  // +key*E3
    const _Float16* vtb   = vt + ((size_t)(b * HEADS + h) * HEAD_DIM) * SEQ;

    // Q fragments (two 16x32), contiguous 16B runs from global
    v16h qa[2];
    {
        const _Float16* qrow = qkv_h + (base + qt * 16 + row) * E3 + h * HEAD_DIM;
#pragma unroll
        for (int c = 0; c < 2; ++c) {
            h8 lo = *(const h8*)(qrow + c * 32 + hi * 16);
            h8 hp = *(const h8*)(qrow + c * 32 + hi * 16 + 8);
#pragma unroll
            for (int i = 0; i < 8; ++i) { qa[c][i] = lo[i]; qa[c][8 + i] = hp[i]; }
        }
    }

    v8f o[4] = {};
    float m_run[8], l_run[8];
#pragma unroll
    for (int r = 0; r < 8; ++r) { m_run[r] = -INFINITY; l_run[r] = 0.0f; }

#if USE_TDM
    // Prologue: DMA first K/V tiles into buffer 0.
    tdm_load_2d(kbase, LDS_K_OFF, 64, 32, 64, 32, (unsigned long long)E3);
    tdm_load_2d(vtb,   LDS_V_OFF, 32, 64, 32, 64, (unsigned long long)SEQ);
#endif
    int buf = 0;

    for (int kv = 0; kv < SEQ; kv += 32) {
#if USE_TDM
        const bool has_next = (kv + 32) < SEQ;
        if (has_next) {
            // protect in-flight DS reads of the buffer we are about to overwrite
            asm volatile("s_wait_dscnt 0" ::: "memory");
            tdm_load_2d(kbase + (size_t)(kv + 32) * E3,
                        LDS_K_OFF + (buf ^ 1) * 4096, 64, 32, 64, 32,
                        (unsigned long long)E3);
            tdm_load_2d(vtb + kv + 32,
                        LDS_V_OFF + (buf ^ 1) * 4096, 32, 64, 32, 64,
                        (unsigned long long)SEQ);
            __builtin_amdgcn_s_wait_tensorcnt(2);   // current buffer's 2 DMAs done
        } else {
            __builtin_amdgcn_s_wait_tensorcnt(0);
        }
        const _Float16* kbuf = ldsK + buf * (32 * 64);
        const _Float16* vbuf = ldsV + buf * (64 * 32);
#endif

        // ---- scores: two 16x16 tiles ----
        v8f s0 = {}, s1 = {};
#pragma unroll
        for (int t = 0; t < 2; ++t) {
            v8f sc = {};
#pragma unroll
            for (int c = 0; c < 2; ++c) {
#if USE_TDM
                v16h kb = *(const v16h*)(kbuf + (t * 16 + row) * 64 + c * 32 + hi * 16);
#else
                v16h kb = *(const v16h*)(kbase + (size_t)(kv + t * 16 + row) * E3 +
                                         c * 32 + hi * 16);
#endif
                sc = wmma_f16(qa[c], kb, sc);
            }
            if (t == 0) s0 = sc; else s1 = sc;
        }

        // ---- online softmax ----
        float esc_arr[8];
#pragma unroll
        for (int r = 0; r < 8; ++r) {
            float a0 = s0[r] * SM, a1 = s1[r] * SM;
            float mx = fmaxf(a0, a1);
#pragma unroll
            for (int msk = 1; msk < 16; msk <<= 1)
                mx = fmaxf(mx, __shfl_xor(mx, msk, 16));
            float mn = fmaxf(m_run[r], mx);
            float p0 = __expf(a0 - mn), p1 = __expf(a1 - mn);
            float rs = p0 + p1;
#pragma unroll
            for (int msk = 1; msk < 16; msk <<= 1)
                rs += __shfl_xor(rs, msk, 16);
            float esc = __expf(m_run[r] - mn);
            l_run[r] = l_run[r] * esc + rs;
            m_run[r] = mn;
            esc_arr[r] = esc;
            lds_p[(r + 8 * hi) * 34 + row]      = (_Float16)p0;
            lds_p[(r + 8 * hi) * 34 + 16 + row] = (_Float16)p1;
        }
#pragma unroll
        for (int g = 0; g < 4; ++g)
#pragma unroll
            for (int r = 0; r < 8; ++r)
                o[g][r] *= esc_arr[r];

        asm volatile("s_wait_dscnt 0" ::: "memory");

        // P as 16x32 A-fragment from LDS
        v16h pa;
#pragma unroll
        for (int i = 0; i < 16; ++i) {
            int kk = hi * 16 + i;
            pa[i] = lds_p[row * 34 + kk];
        }

        // ---- PV ----
#pragma unroll
        for (int g = 0; g < 4; ++g) {
            const int d = g * 16 + row;
#if USE_TDM
            v16h vb = *(const v16h*)(vbuf + d * 32 + hi * 16);
#else
            v16h vb = *(const v16h*)(vtb + (size_t)d * SEQ + kv + hi * 16);
#endif
            o[g] = wmma_f16(pa, vb, o[g]);
        }
        buf ^= 1;
    }

#pragma unroll
    for (int g = 0; g < 4; ++g)
#pragma unroll
        for (int r = 0; r < 8; ++r) {
            float inv = 1.0f / l_run[r];
            attn_h[(base + qt * 16 + r + 8 * hi) * EMBED + h * HEAD_DIM + g * 16 + row] =
                (_Float16)(o[g][r] * inv);
        }
}

// ---------------------------------------------------------------------------
extern "C" void kernel_launch(void* const* d_in, const int* in_sizes, int n_in,
                              void* d_out, int out_size, void* d_ws, size_t ws_size,
                              hipStream_t stream) {
    const float* x    = (const float*)d_in[0];
    const float* Wqkv = (const float*)d_in[1];
    const float* bqkv = (const float*)d_in[2];
    const float* Wout = (const float*)d_in[3];
    const float* bout = (const float*)d_in[4];
    float* out = (float*)d_out;

    _Float16* ws     = (_Float16*)d_ws;
    _Float16* xh     = ws;                                   //  8 MB
    _Float16* WqkvT  = xh     + (size_t)ROWS * EMBED;        //  6 MB
    _Float16* WoutT  = WqkvT  + (size_t)E3 * EMBED;          //  2 MB
    _Float16* qkv_h  = WoutT  + (size_t)EMBED * EMBED;       // 24 MB
    _Float16* vt     = qkv_h  + (size_t)ROWS * E3;           //  8 MB
    _Float16* attn_h = vt     + (size_t)BATCH * HEADS * HEAD_DIM * SEQ;  // 8 MB

    {
        int n = ROWS * EMBED;
        cvt_f32_to_f16<<<(n + 255) / 256, 256, 0, stream>>>(x, xh, n);
    }
    {
        size_t n = (size_t)EMBED * E3;
        transpose_cvt<<<(unsigned)((n + 255) / 256), 256, 0, stream>>>(Wqkv, WqkvT, EMBED, E3);
    }
    {
        size_t n = (size_t)EMBED * EMBED;
        transpose_cvt<<<(unsigned)((n + 255) / 256), 256, 0, stream>>>(Wout, WoutT, EMBED, EMBED);
    }

    gemm_bias_wmma<_Float16><<<dim3(E3 / 64, ROWS / 32), 32, 0, stream>>>(
        xh, WqkvT, bqkv, qkv_h, ROWS, E3, EMBED);

    {
        size_t n = (size_t)BATCH * HEADS * HEAD_DIM * SEQ;
        build_vt<<<(unsigned)((n + 255) / 256), 256, 0, stream>>>(qkv_h, vt);
    }

    flash_attn_wmma<<<dim3(SEQ / 16, HEADS, BATCH), 32, LDS_TOTAL, stream>>>(
        qkv_h, vt, attn_h);

    gemm_bias_wmma<float><<<dim3(EMBED / 64, ROWS / 32), 32, 0, stream>>>(
        attn_h, WoutT, bout, out, ROWS, EMBED, EMBED);
}
